// GatedNet_34995393528527
// MI455X (gfx1250) — compile-verified, matrix operand order checked by reading, not run on hardware
//
#include <hip/hip_runtime.h>
#include <hip/hip_bf16.h>
#include <math.h>

typedef __attribute__((ext_vector_type(16))) __bf16 v16bf;
typedef __attribute__((ext_vector_type(8)))  float  v8f;

#define TILE_M 256      // 8 waves x 32 rows
#define TILE_N 64
#define TILE_K 32
#define ASTRIDE_U 17    // uints per LDS row (68 B -> odd bank stride, conflict-friendly)

__device__ __forceinline__ unsigned short f2bf(float f) {
    // round-to-nearest-even fp32 -> bf16
    unsigned int u = __float_as_uint(f);
    unsigned int r = (u + 0x7fffu + ((u >> 16) & 1u)) >> 16;
    return (unsigned short)r;
}

__device__ __forceinline__ unsigned int pack2(float lo, float hi) {
    return (unsigned int)f2bf(lo) | ((unsigned int)f2bf(hi) << 16);
}

__device__ __forceinline__ float bf2f(unsigned int us) {
    return __uint_as_float(us << 16);
}

union Frag {
    unsigned int u[8];
    v16bf v;
};

__device__ __forceinline__ void atomic_add_f32(float* p, float v) {
    __hip_atomic_fetch_add(p, v, __ATOMIC_RELAXED, __HIP_MEMORY_SCOPE_AGENT);
}

// Fragment load honoring the 16-bit A/B WMMA layout:
// lanes 0-15 hold K {0..7, 16..23}; lanes 16-31 hold K {8..15, 24..31}.
__device__ __forceinline__ v16bf load_frag(const unsigned int* S, int rowBase, int ml, int half) {
    Frag f;
    int base = (rowBase + ml) * ASTRIDE_U + half * 4;
    #pragma unroll
    for (int j = 0; j < 4; ++j) { f.u[j] = S[base + j]; f.u[4 + j] = S[base + 8 + j]; }
    return f.v;
}

// C[M x Nc] = act(A[M x K] @ B + bias)
// A row-major fp32. btrans==0: B is [K x Nc] row-major; btrans==1: B is [Nc x K] (use B^T).
// act: 0 = none, 1 = relu.  obf16: 0 -> fp32 out, 1 -> packed bf16 out.
__global__ __launch_bounds__(256)
void gemm_bias_act_wmma(const float* __restrict__ A, const float* __restrict__ B,
                        const float* __restrict__ bias, void* __restrict__ Cout,
                        int M, int Nc, int K, int btrans, int act, int obf16) {
    __shared__ unsigned int As_u[TILE_M * ASTRIDE_U];   // 256x32 bf16 packed: 17408 B
    __shared__ unsigned int Bs_u[TILE_N * ASTRIDE_U];   // 64x32 bf16 packed (Bs[n][k]): 4352 B

    const int tid  = threadIdx.x;
    const int wave = tid >> 5;      // 0..7 -> 32-row strip
    const int lane = tid & 31;
    const int half = lane >> 4;
    const int ml   = lane & 15;
    const int m0   = blockIdx.x * TILE_M;
    const int n0   = blockIdx.y * TILE_N;

    const bool kAligned = ((K & 3) == 0);

    v8f acc[2][4] = {};

    for (int k0 = 0; k0 < K; k0 += TILE_K) {
        const bool fullK = (k0 + TILE_K <= K);
        // ================= stage A tile: 256 rows x 32 k =================
        {
            const int kk = (tid & 1) * 16;
            const bool fullM = (m0 + TILE_M <= M);
            #pragma unroll
            for (int p = 0; p < 2; ++p) {
                const int row = (tid >> 1) + p * 128;
                const int gr  = m0 + row;
                unsigned int* d = &As_u[row * ASTRIDE_U + (kk >> 1)];
                const float* s = A + (size_t)gr * K + k0 + kk;
                if (fullM && fullK && kAligned) {
                    // vector fast path: global_load_b128 x4
                    if (k0 + TILE_K < K) __builtin_prefetch(s + TILE_K, 0, 0);
                    #pragma unroll
                    for (int j = 0; j < 4; ++j) {
                        float4 f = ((const float4*)s)[j];
                        d[2 * j]     = pack2(f.x, f.y);
                        d[2 * j + 1] = pack2(f.z, f.w);
                    }
                } else if (fullM && fullK) {
                    // unguarded scalar (K not 16B-aligned, e.g. K=67 interior tiles)
                    #pragma unroll
                    for (int j = 0; j < 8; ++j)
                        d[j] = pack2(s[2 * j], s[2 * j + 1]);
                } else {
                    // guarded tail path (rare)
                    #pragma unroll
                    for (int j = 0; j < 8; ++j) {
                        int k = k0 + kk + 2 * j;
                        float f0 = (gr < M && k     < K) ? A[(size_t)gr * K + k]     : 0.0f;
                        float f1 = (gr < M && k + 1 < K) ? A[(size_t)gr * K + k + 1] : 0.0f;
                        d[j] = pack2(f0, f1);
                    }
                }
            }
        }
        // ================= stage B tile as Bs[n][k]: 64 x 32 =================
        {
            const int n  = tid >> 2;
            const int kk = (tid & 3) * 8;
            const int gn = n0 + n;
            unsigned int* d = &Bs_u[n * ASTRIDE_U + (kk >> 1)];
            const bool fullB = (n0 + TILE_N <= Nc) && fullK;
            if (fullB && btrans && kAligned) {
                const float* s = B + (size_t)gn * K + k0 + kk;
                #pragma unroll
                for (int j = 0; j < 2; ++j) {
                    float4 f = ((const float4*)s)[j];
                    d[2 * j]     = pack2(f.x, f.y);
                    d[2 * j + 1] = pack2(f.z, f.w);
                }
            } else if (fullB && btrans) {
                const float* s = B + (size_t)gn * K + k0 + kk;
                #pragma unroll
                for (int j = 0; j < 4; ++j)
                    d[j] = pack2(s[2 * j], s[2 * j + 1]);
            } else if (fullB) {
                #pragma unroll
                for (int j = 0; j < 4; ++j) {
                    int k = k0 + kk + 2 * j;
                    d[j] = pack2(B[(size_t)k * Nc + gn], B[(size_t)(k + 1) * Nc + gn]);
                }
            } else {
                #pragma unroll
                for (int j = 0; j < 4; ++j) {
                    int k = k0 + kk + 2 * j;
                    float f0 = 0.0f, f1 = 0.0f;
                    if (gn < Nc && k < K)
                        f0 = btrans ? B[(size_t)gn * K + k] : B[(size_t)k * Nc + gn];
                    if (gn < Nc && k + 1 < K)
                        f1 = btrans ? B[(size_t)gn * K + k + 1] : B[(size_t)(k + 1) * Nc + gn];
                    d[j] = pack2(f0, f1);
                }
            }
        }
        __syncthreads();

        // ================= compute: 2 A-frags x 4 B-frags = 8 WMMAs =================
        v16bf a0 = load_frag(As_u, wave * 32,      ml, half);
        v16bf a1 = load_frag(As_u, wave * 32 + 16, ml, half);
        #pragma unroll
        for (int nb = 0; nb < 4; ++nb) {
            v16bf bv = load_frag(Bs_u, nb * 16, ml, half);
            acc[0][nb] = __builtin_amdgcn_wmma_f32_16x16x32_bf16(
                false, a0, false, bv, (short)0, acc[0][nb], false, false);
            acc[1][nb] = __builtin_amdgcn_wmma_f32_16x16x32_bf16(
                false, a1, false, bv, (short)0, acc[1][nb], false, false);
        }
        __syncthreads();
    }

    // ---- epilogue: C layout lanes 0-15: (M=r, N=lane); lanes 16-31: (M=8+r, N=lane-16)
    #pragma unroll
    for (int s = 0; s < 2; ++s) {
        #pragma unroll
        for (int nb = 0; nb < 4; ++nb) {
            int col = n0 + nb * 16 + ml;
            if (col >= Nc) continue;
            float bv = bias ? bias[col] : 0.0f;
            #pragma unroll
            for (int r = 0; r < 8; ++r) {
                int row = m0 + wave * 32 + s * 16 + half * 8 + r;
                if (row >= M) continue;
                float v = acc[s][nb][r] + bv;
                if (act) v = fmaxf(v, 0.0f);
                if (obf16)
                    ((unsigned short*)Cout)[(size_t)row * Nc + col] = f2bf(v);
                else
                    ((float*)Cout)[(size_t)row * Nc + col] = v;
            }
        }
    }
}

__global__ void concat_kernel(const float* __restrict__ x, const float* __restrict__ pos,
                              float* __restrict__ hcat, int N) {
    long long i = (long long)blockIdx.x * blockDim.x + threadIdx.x;
    if (i >= (long long)N * 67) return;
    int n = (int)(i / 67), k = (int)(i % 67);
    hcat[i] = (k < 64) ? x[(size_t)n * 64 + k] : pos[(size_t)n * 3 + (k - 64)];
}

__global__ void fill0_kernel(float* __restrict__ p, long long n) {
    long long i = (long long)blockIdx.x * blockDim.x + threadIdx.x;
    if (i < n) p[i] = 0.0f;
}

__global__ void relu_kernel(float* __restrict__ p, long long n) {
    long long i = (long long)blockIdx.x * blockDim.x + threadIdx.x;
    if (i < n) p[i] = fmaxf(p[i], 0.0f);
}

// one wave32 per edge: gather bf16 m[src] row (4 elems = b64 per lane),
// expand to fp32, scatter-add into fp32 agg[dst] with hardware atomics.
__global__ __launch_bounds__(256)
void edge_scatter_bf16_kernel(const unsigned short* __restrict__ mbf,
                              const int* __restrict__ src, const int* __restrict__ dst,
                              float* __restrict__ agg, int E) {
    int e = blockIdx.x * 8 + (threadIdx.x >> 5);
    int lane = threadIdx.x & 31;
    if (e >= E) return;
    int s = src[e], d = dst[e];
    const uint2 w = *(const uint2*)&mbf[(size_t)s * 128 + lane * 4];   // global_load_b64
    float* p = &agg[(size_t)d * 128 + lane * 4];
    atomic_add_f32(&p[0], bf2f(w.x & 0xffffu));
    atomic_add_f32(&p[1], bf2f(w.x >> 16));
    atomic_add_f32(&p[2], bf2f(w.y & 0xffffu));
    atomic_add_f32(&p[3], bf2f(w.y >> 16));
}

// torch GRUCell: gates [r, z, n]; h updated in place
__global__ void gru_kernel(const float* __restrict__ gi, const float* __restrict__ gh,
                           float* __restrict__ h, int N) {
    long long i = (long long)blockIdx.x * blockDim.x + threadIdx.x;
    if (i >= (long long)N * 128) return;
    int n = (int)(i >> 7), f = (int)(i & 127);
    size_t b = (size_t)n * 384;
    float ir = gi[b + f], iz = gi[b + 128 + f], inn = gi[b + 256 + f];
    float hr = gh[b + f], hz = gh[b + 128 + f], hn  = gh[b + 256 + f];
    float hv = h[i];
    float r  = 1.0f / (1.0f + __expf(-(ir + hr)));
    float z  = 1.0f / (1.0f + __expf(-(iz + hz)));
    float nn = tanhf(inn + r * hn);
    h[i] = (1.0f - z) * nn + z * hv;
}

// per-block LDS accumulation over 64 graphs x 16 outs, then global atomics
__global__ __launch_bounds__(256)
void readout_kernel(const float* __restrict__ h2, const int* __restrict__ batch,
                    float* __restrict__ out, int N) {
    __shared__ float sacc[64 * 16];
    int tid = threadIdx.x;
    for (int i = tid; i < 64 * 16; i += 256) sacc[i] = 0.0f;
    __syncthreads();
    int base = blockIdx.x * 256;
    int o = tid & 15;
    #pragma unroll
    for (int it = 0; it < 16; ++it) {
        int n = base + it * 16 + (tid >> 4);
        if (n < N) {
            int g = batch[n];
            atomicAdd(&sacc[g * 16 + o], h2[(size_t)n * 16 + o]);   // ds_add_f32
        }
    }
    __syncthreads();
    for (int i = tid; i < 64 * 16; i += 256)
        atomic_add_f32(&out[i], sacc[i]);
}

extern "C" void kernel_launch(void* const* d_in, const int* in_sizes, int n_in,
                              void* d_out, int out_size, void* d_ws, size_t ws_size,
                              hipStream_t stream) {
    (void)n_in; (void)ws_size;
    const float* x       = (const float*)d_in[0];
    const float* pos     = (const float*)d_in[1];
    const int*   eidx    = (const int*)d_in[2];
    const int*   batch   = (const int*)d_in[3];
    const float* node_w1 = (const float*)d_in[4];
    const float* node_b1 = (const float*)d_in[5];
    const float* node_w2 = (const float*)d_in[6];
    const float* node_b2 = (const float*)d_in[7];
    const float* conv_w  = (const float*)d_in[8];
    const float* gru_wih = (const float*)d_in[9];
    const float* gru_whh = (const float*)d_in[10];
    const float* gru_bih = (const float*)d_in[11];
    const float* gru_bhh = (const float*)d_in[12];
    const float* lin1_w  = (const float*)d_in[13];
    const float* lin1_b  = (const float*)d_in[14];
    const float* lin2_w  = (const float*)d_in[15];
    const float* lin2_b  = (const float*)d_in[16];

    const int N = in_sizes[0] / 64;   // CIN = 64
    const int E = in_sizes[2] / 2;    // edge_index is [2, E]
    const int H = 128, L = 3, KCAT = 67;

    const int* src  = eidx;
    const int* dstE = eidx + E;

    // workspace layout (fp32), with reuse
    float* ws = (float*)d_ws;
    size_t o = 0;
    float* h   = ws + o; o += (size_t)N * H;        // node state
    float* m   = ws + o; o += (size_t)N * H;        // mlp tmp / lin1 out / bf16 messages
    float* agg = ws + o; o += (size_t)N * H;        // aggregated messages / lin2 out
    float* gi  = ws + o; o += (size_t)N * 3 * H;    // GRU input gates / hcat
    float* gh  = ws + o; o += (size_t)N * 3 * H;    // GRU hidden gates
    float* hcat = gi;                   // N*67  < N*384
    float* tmp  = m;
    float* h1   = m;                    // N*64
    float* h2   = agg;                  // N*16
    unsigned short* mbf = (unsigned short*)m;   // bf16 messages, N*128 ushorts

    dim3 blk(256);
    auto cdiv = [](long long a, long long b) { return (unsigned)((a + b - 1) / b); };

    // node MLP: h = relu(concat(x,pos) @ W1 + b1) @ W2 + b2
    concat_kernel<<<cdiv((long long)N * KCAT, 256), blk, 0, stream>>>(x, pos, hcat, N);
    gemm_bias_act_wmma<<<dim3(cdiv(N, TILE_M), cdiv(H, TILE_N)), blk, 0, stream>>>(
        hcat, node_w1, node_b1, tmp, N, H, KCAT, 0, 1, 0);
    gemm_bias_act_wmma<<<dim3(cdiv(N, TILE_M), cdiv(H, TILE_N)), blk, 0, stream>>>(
        tmp, node_w2, node_b2, h, N, H, H, 0, 0, 0);

    for (int l = 0; l < L; ++l) {
        for (int i = 0; i < L; ++i) {
            // m = h @ conv_w[l, i]  -> emitted directly as bf16 (gather bandwidth / L2 footprint)
            gemm_bias_act_wmma<<<dim3(cdiv(N, TILE_M), cdiv(H, TILE_N)), blk, 0, stream>>>(
                h, conv_w + (size_t)(l * L + i) * H * H, nullptr, mbf, N, H, H, 0, 0, 1);
            // agg = segment_sum(m[src] -> dst), fp32 accumulation
            fill0_kernel<<<cdiv((long long)N * H, 256), blk, 0, stream>>>(agg, (long long)N * H);
            edge_scatter_bf16_kernel<<<cdiv(E, 8), blk, 0, stream>>>(mbf, src, dstE, agg, E);
            // GRU gates: gi = agg @ wih^T + bih ; gh = h @ whh^T + bhh
            gemm_bias_act_wmma<<<dim3(cdiv(N, TILE_M), cdiv(3 * H, TILE_N)), blk, 0, stream>>>(
                agg, gru_wih + (size_t)l * 3 * H * H, gru_bih + (size_t)l * 3 * H,
                gi, N, 3 * H, H, 1, 0, 0);
            gemm_bias_act_wmma<<<dim3(cdiv(N, TILE_M), cdiv(3 * H, TILE_N)), blk, 0, stream>>>(
                h, gru_whh + (size_t)l * 3 * H * H, gru_bhh + (size_t)l * 3 * H,
                gh, N, 3 * H, H, 1, 0, 0);
            gru_kernel<<<cdiv((long long)N * H, 256), blk, 0, stream>>>(gi, gh, h, N);
        }
        relu_kernel<<<cdiv((long long)N * H, 256), blk, 0, stream>>>(h, (long long)N * H);
    }

    // head: h1 = relu(h @ lin1 + b); h2 = h1 @ lin2 + b; out = segment_sum(h2, batch)
    gemm_bias_act_wmma<<<dim3(cdiv(N, TILE_M), cdiv(64, TILE_N)), blk, 0, stream>>>(
        h, lin1_w, lin1_b, h1, N, 64, H, 0, 1, 0);
    gemm_bias_act_wmma<<<dim3(cdiv(N, TILE_M), cdiv(16, TILE_N)), blk, 0, stream>>>(
        h1, lin2_w, lin2_b, h2, N, 16, 64, 0, 0, 0);

    fill0_kernel<<<cdiv(out_size, 256), blk, 0, stream>>>((float*)d_out, out_size);
    readout_kernel<<<cdiv(N, 256), blk, 0, stream>>>(h2, batch, (float*)d_out, N);
}
